// QuantumAutoEncoder_89867895701967
// MI455X (gfx1250) — matrix operand order repaired and checked by reading, not compiled, not run on hardware
//
#include <hip/hip_runtime.h>
#include <hip/hip_bf16.h>

// ---------------------------------------------------------------------------
// Types for CDNA5 WMMA (gfx1250, wave32)
// ---------------------------------------------------------------------------
typedef __attribute__((ext_vector_type(16))) __bf16        v16bf;
typedef __attribute__((ext_vector_type(8)))  float         v8f;
typedef __attribute__((ext_vector_type(4)))  unsigned int  v4u;

union FragBF {
    v16bf v;
    v4u   q[2];
};

__device__ __forceinline__ unsigned short f2bf(float f) {
    unsigned int u = __float_as_uint(f);
    u += 0x7FFFu + ((u >> 16) & 1u);       // round-to-nearest-even
    return (unsigned short)(u >> 16);
}

// CDNA5 async global->LDS 16B copy (ASYNCcnt-tracked), via inline asm.
__device__ __forceinline__ void async_copy16(unsigned int lds_off, unsigned long long gaddr) {
    asm volatile("global_load_async_to_lds_b128 %0, %1, off"
                 :: "v"(lds_off), "v"(gaddr)
                 : "memory");
}

// ---------------------------------------------------------------------------
// GEMM: C[M,N] = A[M,KP] * Bt[N,KP]^T + bias[N]
//   A  : bf16 [M, KP]  row-major (KP % 64 == 0, zero-padded)
//   Bt : bf16 [Np, KP] transposed weights (Np covers full 128-col tiles)
// Block tile 128x128, K-step 64. 256 threads = 8 waves in 4(M) x 2(N) grid;
// each wave owns 32x64 = 2x4 WMMA tiles, 16 v_wmma per K-step.
// Double-buffered 64KB LDS fed by async global->LDS copies; steady-state loop
// is branch-free (last iteration peeled, KP is a template constant).
// ---------------------------------------------------------------------------
#define BM 128
#define BN 128
#define BK 64

template <bool OUT_BF16, int KP>
__global__ __launch_bounds__(256)
void gemm_bf16wmma(const unsigned short* __restrict__ A,
                   const unsigned short* __restrict__ Bt,
                   const float* __restrict__ bias,
                   void* __restrict__ Cv,
                   int N) {
    // Layout (bytes): [buf0 A 16K][buf0 B 16K][buf1 A 16K][buf1 B 16K]
    __shared__ unsigned short smem[4 * BM * BK / 2 * 2];   // 64 KB

    const int tid  = threadIdx.x;
    const int lane = tid & 31;
    const int wave = tid >> 5;
    const int wm   = wave & 3;               // 0..3 : 32-row slab
    const int wn   = wave >> 2;              // 0..1 : 64-col slab
    const int blockN = blockIdx.x * BN;
    const int blockM = blockIdx.y * BM;

    const int l15 = lane & 15;
    const int hiA = (lane >= 16) ? 4 : 0;    // A-frag dword skew (K +8 half)
    const int hiB = (lane >= 16) ? 4 : 0;    // B-frag dword skew (K +8 half)

    v8f acc[2][4];
    #pragma unroll
    for (int i = 0; i < 2; ++i)
        #pragma unroll
        for (int j = 0; j < 4; ++j)
            acc[i][j] = (v8f){0.f, 0.f, 0.f, 0.f, 0.f, 0.f, 0.f, 0.f};

    // ---- per-thread async-copy chunk addresses (4 x 16B per matrix/tile) ---
    // Tile = 128 rows x 128 bytes -> 1024 x 16B chunks, 4 per thread.
    const unsigned int smem_base = (unsigned int)(uintptr_t)&smem[0];
    unsigned int       ldsA[4];
    unsigned long long gA[4], gB[4];
    #pragma unroll
    for (int i = 0; i < 4; ++i) {
        const int c   = tid + 256 * i;
        const int row = c >> 3;              // 0..127
        const int seg = c & 7;               // 16B segment within 128B row
        ldsA[i] = smem_base + row * 128 + seg * 16;
        gA[i] = (unsigned long long)(uintptr_t)A +
                ((unsigned long long)(blockM + row) * KP + seg * 8) * 2;
        gB[i] = (unsigned long long)(uintptr_t)Bt +
                ((unsigned long long)(blockN + row) * KP + seg * 8) * 2;
    }

    auto issue_tile = [&](unsigned int bufofs) {
        #pragma unroll
        for (int i = 0; i < 4; ++i) {
            async_copy16(ldsA[i] + bufofs,           gA[i]);
            async_copy16(ldsA[i] + bufofs + 0x4000,  gB[i]);
            gA[i] += 2 * BK;                 // advance 128 bytes (one K-tile)
            gB[i] += 2 * BK;
        }
    };

    auto compute = [&](int buf) {
        const unsigned int* pA = (const unsigned int*)&smem[(size_t)buf * 16384];
        const unsigned int* pB = pA + 4096;
        #pragma unroll
        for (int kk2 = 0; kk2 < 32; kk2 += 16) {   // two 16x16x32 K-chunks
            FragBF afr[2], bfr[4];
            #pragma unroll
            for (int i = 0; i < 2; ++i) {
                const int base = (wm * 32 + i * 16 + l15) * 32 + kk2 + hiA;
                afr[i].q[0] = *(const v4u*)(pA + base);
                afr[i].q[1] = *(const v4u*)(pA + base + 8);
            }
            #pragma unroll
            for (int j = 0; j < 4; ++j) {
                const int base = (wn * 64 + j * 16 + l15) * 32 + kk2 + hiB;
                bfr[j].q[0] = *(const v4u*)(pB + base);
                bfr[j].q[1] = *(const v4u*)(pB + base + 8);
            }
            #pragma unroll
            for (int i = 0; i < 2; ++i)
                #pragma unroll
                for (int j = 0; j < 4; ++j)
                    acc[i][j] = __builtin_amdgcn_wmma_f32_16x16x32_bf16(
                        false, afr[i].v, false, bfr[j].v,
                        (short)0, acc[i][j], false, false);
        }
    };

    constexpr int NT = KP / BK;
    issue_tile(0);
    for (int t = 0; t < NT - 1; ++t) {
        issue_tile((unsigned int)((t + 1) & 1) * 0x8000u);
        // 8 in-flight copies belong to tile t+1; retire tile t's 8 (in-order).
        asm volatile("s_wait_asynccnt 0x8" ::: "memory");
        __syncthreads();
        compute(t & 1);
        __syncthreads();
    }
    asm volatile("s_wait_asynccnt 0x0" ::: "memory");
    __syncthreads();
    compute((NT - 1) & 1);

    // Epilogue: C/D layout (lanes 0-15: M=r, lanes 16-31: M=r+8).
    const int rofs = (lane >= 16) ? 8 : 0;
    float* Cf = (float*)Cv;
    unsigned short* Ch = (unsigned short*)Cv;
    #pragma unroll
    for (int j = 0; j < 4; ++j) {
        const int col = blockN + wn * 64 + j * 16 + l15;
        if (col < N) {
            const float bv = bias[col];
            #pragma unroll
            for (int i = 0; i < 2; ++i) {
                #pragma unroll
                for (int r = 0; r < 8; ++r) {
                    const int row = blockM + wm * 32 + i * 16 + rofs + r;
                    const float v = acc[i][j][r] + bv;
                    if (OUT_BF16) Ch[(size_t)row * N + col] = f2bf(v);
                    else          Cf[(size_t)row * N + col] = v;
                }
            }
        }
    }
}

// ---------------------------------------------------------------------------
// fp32 [R,C] -> bf16 [R,Cp] with zero pad (row-K padding for GEMM A operands)
// ---------------------------------------------------------------------------
__global__ __launch_bounds__(256)
void cvt_pad_bf16(const float* __restrict__ X, unsigned short* __restrict__ Y,
                  int C, int Cp, int total) {
    const int idx = blockIdx.x * 256 + threadIdx.x;
    if (idx >= total) return;
    const int r = idx / Cp, c = idx - r * Cp;
    Y[idx] = (c < C) ? f2bf(X[(size_t)r * C + c]) : (unsigned short)0;
}

// ---------------------------------------------------------------------------
// fp32 W[K,N] -> bf16 Wt[Np,Kp] transposed, zero-padded (LDS-tiled transpose)
// ---------------------------------------------------------------------------
__global__ __launch_bounds__(256)
void cvt_transpose_bf16(const float* __restrict__ W, unsigned short* __restrict__ Wt,
                        int K, int N, int Kp, int Np) {
    __shared__ float tile[32][33];
    const int tid = threadIdx.x;
    const int lx = tid & 31, ly = tid >> 5;       // 32 x 8
    const int kt = blockIdx.x * 32, nt = blockIdx.y * 32;
    #pragma unroll
    for (int i = 0; i < 4; ++i) {
        const int k = kt + ly + i * 8, n = nt + lx;
        tile[ly + i * 8][lx] = (k < K && n < N) ? W[(size_t)k * N + n] : 0.f;
    }
    __syncthreads();
    #pragma unroll
    for (int i = 0; i < 4; ++i) {
        const int n = nt + ly + i * 8, k = kt + lx;
        if (n < Np && k < Kp)
            Wt[(size_t)n * Kp + k] = f2bf(tile[lx][ly + i * 8]);
    }
}

// ---------------------------------------------------------------------------
// 10-qubit state-vector circuit (AmplitudeEmbedding + StronglyEntanglingLayers
// + probs). One 256-thread block per batch row; state in LDS. Emits bf16
// probabilities (consumed by the next WMMA GEMM).
// ---------------------------------------------------------------------------
#define NQB 10
#define NLAY 3
#define DIM 1024

__global__ __launch_bounds__(256)
void qcircuit(const float* __restrict__ X, const float* __restrict__ W,
              unsigned short* __restrict__ Out) {
    __shared__ float re[DIM];
    __shared__ float im[DIM];
    __shared__ float red[256];

    const int t = threadIdx.x;
    const size_t base = (size_t)blockIdx.x * DIM;

    float ss = 0.f;
    #pragma unroll
    for (int i = 0; i < 4; ++i) {
        const float v = X[base + t + 256 * i];
        re[t + 256 * i] = v;
        im[t + 256 * i] = 0.f;
        ss += v * v;
    }
    red[t] = ss;
    __syncthreads();
    for (int s = 128; s > 0; s >>= 1) {
        if (t < s) red[t] += red[t + s];
        __syncthreads();
    }
    const float inv = rsqrtf(red[0]);
    #pragma unroll
    for (int i = 0; i < 4; ++i) re[t + 256 * i] *= inv;
    __syncthreads();

    for (int l = 0; l < NLAY; ++l) {
        // Rot(phi, theta, omega) = RZ(omega) RY(theta) RZ(phi) per qubit
        for (int q = 0; q < NQB; ++q) {
            const float* wq = W + (l * NQB + q) * 3;
            const float phi = wq[0], th = wq[1], om = wq[2];
            const float c  = __cosf(0.5f * th), s = __sinf(0.5f * th);
            const float ap = 0.5f * (phi + om), am = 0.5f * (phi - om);
            const float cap = __cosf(ap), sap = __sinf(ap);
            const float cam = __cosf(am), sam = __sinf(am);
            const float u00r =  cap * c, u00i = -sap * c;
            const float u01r = -cam * s, u01i = -sam * s;
            const float u10r =  cam * s, u10i = -sam * s;
            const float u11r =  cap * c, u11i =  sap * c;
            const int bq  = (NQB - 1) - q;      // wire 0 = MSB of flat index
            const int bit = 1 << bq;
            #pragma unroll
            for (int pp = 0; pp < 2; ++pp) {
                const int p  = t + pp * 256;    // 512 pairs, 2 per thread
                const int i0 = ((p >> bq) << (bq + 1)) | (p & (bit - 1));
                const int i1 = i0 | bit;
                const float a0r = re[i0], a0i = im[i0];
                const float a1r = re[i1], a1i = im[i1];
                re[i0] = u00r * a0r - u00i * a0i + u01r * a1r - u01i * a1i;
                im[i0] = u00r * a0i + u00i * a0r + u01r * a1i + u01i * a1r;
                re[i1] = u10r * a0r - u10i * a0i + u11r * a1r - u11i * a1i;
                im[i1] = u10r * a0i + u10i * a0r + u11r * a1i + u11i * a1r;
            }
            __syncthreads();
        }
        // ring CNOTs, range r = l % (NQ-1) + 1
        const int rr = l % (NQB - 1) + 1;
        for (int q = 0; q < NQB; ++q) {
            const int bcb = (NQB - 1) - q;
            const int btb = (NQB - 1) - ((q + rr) % NQB);
            const int lo = bcb < btb ? bcb : btb;
            const int hi = bcb < btb ? btb : bcb;
            const int x  = ((t >> lo) << (lo + 1)) | (t & ((1 << lo) - 1));
            const int y  = ((x >> hi) << (hi + 1)) | (x & ((1 << hi) - 1));
            const int i0 = y | (1 << bcb);                 // control=1, target=0
            const int i1 = i0 | (1 << btb);                // control=1, target=1
            const float tr = re[i0], ti = im[i0];
            re[i0] = re[i1]; im[i0] = im[i1];
            re[i1] = tr;     im[i1] = ti;
            __syncthreads();
        }
    }

    #pragma unroll
    for (int i = 0; i < 4; ++i) {
        const int idx = t + 256 * i;
        const float r2 = re[idx], i2 = im[idx];
        Out[base + idx] = f2bf(r2 * r2 + i2 * i2);
    }
}

// ---------------------------------------------------------------------------
// Row-wise abs + L1 normalize into final output
// ---------------------------------------------------------------------------
__global__ __launch_bounds__(256)
void absnorm(const float* __restrict__ Y, float* __restrict__ Out, int n) {
    __shared__ float red[256];
    const int t = threadIdx.x;
    const size_t base = (size_t)blockIdx.x * n;
    float s = 0.f;
    for (int i = t; i < n; i += 256) s += fabsf(Y[base + i]);
    red[t] = s;
    __syncthreads();
    for (int st = 128; st > 0; st >>= 1) {
        if (t < st) red[t] += red[t + st];
        __syncthreads();
    }
    const float inv = 1.0f / red[0];
    for (int i = t; i < n; i += 256) Out[base + i] = fabsf(Y[base + i]) * inv;
}

// ---------------------------------------------------------------------------
// Pipeline driver
// ---------------------------------------------------------------------------
extern "C" void kernel_launch(void* const* d_in, const int* in_sizes, int n_in,
                              void* d_out, int out_size, void* d_ws, size_t ws_size,
                              hipStream_t stream) {
    (void)in_sizes; (void)n_in; (void)out_size; (void)ws_size;

    const float* inputs = (const float*)d_in[0];
    const float* wEnc   = (const float*)d_in[1];
    const float* wDec   = (const float*)d_in[2];
    const float* W0 = (const float*)d_in[3];
    const float* b0 = (const float*)d_in[4];
    const float* W1 = (const float*)d_in[5];
    const float* b1 = (const float*)d_in[6];
    const float* W2 = (const float*)d_in[7];
    const float* b2 = (const float*)d_in[8];
    const float* W3 = (const float*)d_in[9];
    const float* b3 = (const float*)d_in[10];

    const int Mb = 4096, E = 1024, E2 = 2048, IN = 1000;

    // Workspace layout (256B aligned); ~63 MB total, L2-resident.
    char* w = (char*)d_ws;
    auto alloc = [&](size_t bytes) {
        void* p = (void*)w;
        w += (bytes + 255) & ~(size_t)255;
        return p;
    };
    unsigned short* inbf = (unsigned short*)alloc((size_t)Mb * E  * 2); // inputs, K padded 1000->1024
    unsigned short* W0t  = (unsigned short*)alloc((size_t)E  * E  * 2); // [1024][1024]
    unsigned short* W1t  = (unsigned short*)alloc((size_t)E2 * E  * 2); // [2048][1024]
    unsigned short* W2t  = (unsigned short*)alloc((size_t)E  * E2 * 2); // [1024][2048]
    unsigned short* W3t  = (unsigned short*)alloc((size_t)E  * E  * 2); // [1024][1024] (N padded)
    float*          X0f  = (float*)         alloc((size_t)Mb * E  * 4); // dense0 out (reused as X2f)
    unsigned short* P0bf = (unsigned short*)alloc((size_t)Mb * E  * 2); // enc probs (reused as P1bf)
    unsigned short* X1bf = (unsigned short*)alloc((size_t)Mb * E2 * 2); // dense1 out (region reused for Y3f)
    (void)alloc(256);                                                   // trailing guard
    float*          X2f  = X0f;
    unsigned short* P1bf = P0bf;
    float*          Y3f  = (float*)X1bf;    // 4096*1000*4 <= 4096*2048*2

    const dim3 blk(256);

    // One-time (per launch) bf16 conversion / transposition of operands.
    cvt_pad_bf16<<<dim3((Mb * E + 255) / 256), blk, 0, stream>>>(inputs, inbf, IN, E, Mb * E);
    cvt_transpose_bf16<<<dim3(E / 32,  E / 32),  blk, 0, stream>>>(W0, W0t, IN, E,  E,  E);
    cvt_transpose_bf16<<<dim3(E / 32,  E2 / 32), blk, 0, stream>>>(W1, W1t, E,  E2, E,  E2);
    cvt_transpose_bf16<<<dim3(E2 / 32, E / 32),  blk, 0, stream>>>(W2, W2t, E2, E,  E2, E);
    cvt_transpose_bf16<<<dim3(E / 32,  E / 32),  blk, 0, stream>>>(W3, W3t, E,  IN, E,  E);

    gemm_bf16wmma<false, 1024><<<dim3(E  / BN, Mb / BM), blk, 0, stream>>>(inbf, W0t, b0, X0f, E);
    qcircuit<<<dim3(Mb), blk, 0, stream>>>(X0f, wEnc, P0bf);
    gemm_bf16wmma<true, 1024> <<<dim3(E2 / BN, Mb / BM), blk, 0, stream>>>(P0bf, W1t, b1, X1bf, E2);
    gemm_bf16wmma<false, 2048><<<dim3(E  / BN, Mb / BM), blk, 0, stream>>>(X1bf, W2t, b2, X2f, E);
    qcircuit<<<dim3(Mb), blk, 0, stream>>>(X2f, wDec, P1bf);
    gemm_bf16wmma<false, 1024><<<dim3((IN + BN - 1) / BN, Mb / BM), blk, 0, stream>>>(P1bf, W3t, b3, Y3f, IN);
    absnorm<<<dim3(Mb), blk, 0, stream>>>(Y3f, (float*)d_out, IN);
}
